// DisGNN_82918638617117
// MI455X (gfx1250) — matrix operand order repaired and checked by reading, not compiled, not run on hardware
//
#include <hip/hip_runtime.h>

// ---- types ----
typedef __attribute__((ext_vector_type(16))) __bf16        v16bf;
typedef __attribute__((ext_vector_type(8)))  float         v8f;
typedef __attribute__((ext_vector_type(4)))  unsigned int  u32x4;
typedef __attribute__((ext_vector_type(4)))  float         f32x4;

#define CCH 128      // channels C
#define DED 32       // edge feature dim D
#define KT  9        // K tiles of 32 over z-dim (2C + D = 288)
#define NT  8        // N tiles of 16 over C = 128

__device__ __forceinline__ unsigned short f2bf(float f) {
    unsigned int u = __float_as_uint(f);
    unsigned int r = (u + 0x7FFFu + ((u >> 16) & 1u)) >> 16;
    return (unsigned short)r;
}
__device__ __forceinline__ unsigned int pk2bf(float a, float b) {
    return (unsigned int)f2bf(a) | ((unsigned int)f2bf(b) << 16);
}

// ============================================================
// Layer 1: h = leaky([x | onehot(y[batch])] @ lin_W + lin_b)
// one block (128 threads) per row; x row staged in LDS;
// W access is coalesced across threads for each k.
// ============================================================
__global__ void layer1_kernel(const float* __restrict__ x,
                              const int*   __restrict__ y,
                              const int*   __restrict__ batch,
                              const float* __restrict__ W,    // [C+NC][C]
                              const float* __restrict__ b,    // [C]
                              float*          __restrict__ h,
                              unsigned short* __restrict__ hb,
                              int N) {
    __shared__ float xs[CCH];
    int row = blockIdx.x;
    if (row >= N) return;
    int t = threadIdx.x;                 // 0..127 = output channel
    xs[t] = x[row * CCH + t];
    __syncthreads();
    int cls = y[batch[row]];
    float s = b[t] + W[(CCH + cls) * CCH + t];   // one-hot contribution
    #pragma unroll 8
    for (int k = 0; k < CCH; ++k) s = fmaf(xs[k], W[k * CCH + t], s);
    s = (s >= 0.f) ? s : 0.01f * s;
    h[row * CCH + t]  = s;
    hb[row * CCH + t] = f2bf(s);
}

// ============================================================
// Pre-swizzle the four conv weights [288][128] f32 ->
// B-fragment-major bf16: idx = ((kt*8+nt)*32 + lane)*16 + i
// lane elem i holds W[kt*32 + (lane<16?0:8) + (i&7) + (i<8?0:16)][nt*16 + lane%16]
// ============================================================
__global__ void swizzle_kernel(const float* __restrict__ w0,
                               const float* __restrict__ w1,
                               const float* __restrict__ w2,
                               const float* __restrict__ w3,
                               unsigned short* __restrict__ dstw) {
    const int PER = KT * NT * 32 * 16;   // 36864 per weight
    int idx = blockIdx.x * blockDim.x + threadIdx.x;
    if (idx >= 4 * PER) return;
    int w = idx / PER, r = idx % PER;
    int kt   = r / (NT * 32 * 16);  r %= (NT * 32 * 16);
    int nt   = r >> 9;
    int lane = (r >> 4) & 31;
    int i    = r & 15;
    int k = kt * 32 + ((lane < 16) ? 0 : 8) + (i & 7) + ((i < 8) ? 0 : 16);
    int n = nt * 16 + (lane & 15);
    const float* src = (w == 0) ? w0 : (w == 1) ? w1 : (w == 2) ? w2 : w3;
    dstw[idx] = f2bf(src[k * CCH + n]);
}

// ============================================================
// CGConv edge GEMM + gated message + scatter-add.
// One wave = 16 edges; z = [h[dst] | h[src] | e] (K = 288).
// 9 K-steps x 8 N-tiles x 2 gates = 144 v_wmma per wave.
// ============================================================
union AFrag {
    unsigned int u[8];
    v16bf        v;
};

__global__ void __launch_bounds__(256)
cgconv_kernel(const int*            __restrict__ src,
              const int*            __restrict__ dst,
              const unsigned short* __restrict__ hb,     // [N][128] bf16
              const float*          __restrict__ eattr,  // [E][32] f32
              const unsigned short* __restrict__ Wfsw,   // swizzled bf16
              const unsigned short* __restrict__ Wssw,
              const float*          __restrict__ bf,
              const float*          __restrict__ bs,
              float*                __restrict__ agg,
              int E) {
    int wave = threadIdx.x >> 5;
    int lane = threadIdx.x & 31;
    int tile = blockIdx.x * (blockDim.x >> 5) + wave;
    if (tile * 16 >= E) return;          // wave-uniform

    int e0   = tile * 16;
    int half = lane >> 4;
    int col  = lane & 15;
    int off  = half * 8;                 // A-layout K offset for this half

    int myEdge = e0 + col;               // the A-row this lane feeds
    int sOwn = src[myEdge];
    int dOwn = dst[myEdge];

    v8f accf[NT], accs[NT];
    #pragma unroll
    for (int nt = 0; nt < NT; ++nt)
        #pragma unroll
        for (int r = 0; r < 8; ++r) { accf[nt][r] = 0.f; accs[nt][r] = 0.f; }

    #pragma unroll
    for (int kt = 0; kt < KT; ++kt) {
        AFrag a;
        if (kt < 8) {
            // z columns from h[dst] (kt 0..3) or h[src] (kt 4..7)
            int node = (kt < 4) ? dOwn : sOwn;
            const unsigned short* p = hb + node * CCH + (kt & 3) * 32 + off;
            *(u32x4*)(&a.u[0]) = *(const u32x4*)(p);        // K off..off+7
            *(u32x4*)(&a.u[4]) = *(const u32x4*)(p + 16);   // K off+16..off+23
        } else {
            // edge_attr (f32 -> bf16 on the fly)
            const float* p = eattr + myEdge * DED + off;
            f32x4 f0 = *(const f32x4*)(p);
            f32x4 f1 = *(const f32x4*)(p + 4);
            f32x4 f2 = *(const f32x4*)(p + 16);
            f32x4 f3 = *(const f32x4*)(p + 20);
            a.u[0] = pk2bf(f0.x, f0.y); a.u[1] = pk2bf(f0.z, f0.w);
            a.u[2] = pk2bf(f1.x, f1.y); a.u[3] = pk2bf(f1.z, f1.w);
            a.u[4] = pk2bf(f2.x, f2.y); a.u[5] = pk2bf(f2.z, f2.w);
            a.u[6] = pk2bf(f3.x, f3.y); a.u[7] = pk2bf(f3.z, f3.w);
        }
        #pragma unroll
        for (int nt = 0; nt < NT; ++nt) {
            int fragBase = ((kt * NT + nt) * 32 + lane) * 16;
            AFrag b, c;
            const unsigned short* pf = Wfsw + fragBase;
            *(u32x4*)(&b.u[0]) = *(const u32x4*)(pf);
            *(u32x4*)(&b.u[4]) = *(const u32x4*)(pf + 8);
            accf[nt] = __builtin_amdgcn_wmma_f32_16x16x32_bf16(
                false, a.v, false, b.v, (short)0, accf[nt], false, false);
            const unsigned short* ps = Wssw + fragBase;
            *(u32x4*)(&c.u[0]) = *(const u32x4*)(ps);
            *(u32x4*)(&c.u[4]) = *(const u32x4*)(ps + 8);
            accs[nt] = __builtin_amdgcn_wmma_f32_16x16x32_bf16(
                false, a.v, false, c.v, (short)0, accs[nt], false, false);
        }
    }

    // C/D layout: vgpr r, lane -> edge row M = r + half*8, col = nt*16 + col
    int dstr[8];
    #pragma unroll
    for (int r = 0; r < 8; ++r) dstr[r] = dst[e0 + half * 8 + r];

    #pragma unroll
    for (int nt = 0; nt < NT; ++nt) {
        int n = nt * 16 + col;
        float bfv = bf[n], bsv = bs[n];
        #pragma unroll
        for (int r = 0; r < 8; ++r) {
            float xf = accf[nt][r] + bfv;
            float xg = accs[nt][r] + bsv;
            float sg = 1.f / (1.f + __expf(-xf));
            float sp = (xg > 15.f) ? xg : log1pf(__expf(xg));
            atomicAdd(&agg[dstr[r] * CCH + n], sg * sp);
        }
    }
}

// ============================================================
// h = (leaky?)(h + agg); refresh bf16 copy
// ============================================================
__global__ void finalize_kernel(float* __restrict__ h, unsigned short* __restrict__ hb,
                                const float* __restrict__ agg, int n, int doLeaky) {
    int i = blockIdx.x * blockDim.x + threadIdx.x;
    if (i >= n) return;
    float v = h[i] + agg[i];
    if (doLeaky) v = (v >= 0.f) ? v : 0.01f * v;
    h[i]  = v;
    hb[i] = f2bf(v);
}

// ============================================================
// Mean pool (segment sums via atomics) over [h | onehot]
// ============================================================
__global__ void pool_kernel(const float* __restrict__ h, const int* __restrict__ batch,
                            const int* __restrict__ y, float* __restrict__ sums,
                            float* __restrict__ cnt, int N, int CNC) {
    int i = blockIdx.x * blockDim.x + threadIdx.x;
    int node = i >> 7;
    int c = i & 127;
    if (node >= N) return;
    int g = batch[node];
    atomicAdd(&sums[g * CNC + c], h[node * CCH + c]);
    if (c == 0) {
        atomicAdd(&sums[g * CNC + CCH + y[g]], 1.0f);  // onehot column sum
        atomicAdd(&cnt[g], 1.0f);
    }
}

// ============================================================
// Head: sigmoid(leaky(pooled@fc1+b1)@fc2+b2), one block per graph
// ============================================================
__global__ void head_kernel(const float* __restrict__ sums, const float* __restrict__ cnt,
                            const float* __restrict__ W1, const float* __restrict__ b1,
                            const float* __restrict__ W2, const float* __restrict__ b2,
                            float* __restrict__ out, int CNC) {
    int g = blockIdx.x;
    int t = threadIdx.x;                 // 32 threads
    __shared__ float pooled[256];
    __shared__ float u[32];
    float ic = 1.f / fmaxf(cnt[g], 1.f);
    for (int k = t; k < CNC; k += 32) pooled[k] = sums[g * CNC + k] * ic;
    __syncthreads();
    float s = b1[t];
    for (int k = 0; k < CNC; ++k) s = fmaf(pooled[k], W1[k * 32 + t], s);
    u[t] = (s >= 0.f) ? s : 0.01f * s;
    __syncthreads();
    if (t == 0) {
        float o = b2[0];
        #pragma unroll
        for (int k = 0; k < 32; ++k) o += u[k] * W2[k];
        out[g] = 1.f / (1.f + __expf(-o));
    }
}

// ============================================================
extern "C" void kernel_launch(void* const* d_in, const int* in_sizes, int n_in,
                              void* d_out, int out_size, void* d_ws, size_t ws_size,
                              hipStream_t stream) {
    const float* x      = (const float*)d_in[0];
    const int*   y      = (const int*)  d_in[1];
    const int*   eidx   = (const int*)  d_in[2];
    const float* eattr  = (const float*)d_in[3];
    const int*   batch  = (const int*)  d_in[4];
    const float* lin_W  = (const float*)d_in[5];
    const float* lin_b  = (const float*)d_in[6];
    const float* c1_Wf  = (const float*)d_in[7];
    const float* c1_bf  = (const float*)d_in[8];
    const float* c1_Ws  = (const float*)d_in[9];
    const float* c1_bs  = (const float*)d_in[10];
    const float* c2_Wf  = (const float*)d_in[11];
    const float* c2_bf  = (const float*)d_in[12];
    const float* c2_Ws  = (const float*)d_in[13];
    const float* c2_bs  = (const float*)d_in[14];
    const float* fc1_W  = (const float*)d_in[15];
    const float* fc1_b  = (const float*)d_in[16];
    const float* fc2_W  = (const float*)d_in[17];
    const float* fc2_b  = (const float*)d_in[18];

    const int N   = in_sizes[0] / CCH;      // 50000
    const int E   = in_sizes[3] / DED;      // 800000
    const int G   = in_sizes[1];            // 64
    const int CNC = in_sizes[15] / 32;      // 138 = C + NC

    const int* src = eidx;
    const int* dst = eidx + E;

    // workspace layout
    char* ws = (char*)d_ws;
    size_t o = 0;
    float*          h   = (float*)(ws + o);          o += (size_t)N * CCH * 4;
    unsigned short* hb  = (unsigned short*)(ws + o); o += (size_t)N * CCH * 2;
    float*          agg = (float*)(ws + o);          o += (size_t)N * CCH * 4;
    const int PER = KT * NT * 32 * 16;               // 36864 bf16 per weight
    unsigned short* Wsw = (unsigned short*)(ws + o); o += (size_t)4 * PER * 2;
    float*          sums = (float*)(ws + o);         o += (size_t)G * CNC * 4;
    float*          cnt  = (float*)(ws + o);         o += (size_t)G * 4;
    (void)ws_size; (void)n_in; (void)out_size;

    // layer 1 + weight swizzle (independent)
    layer1_kernel<<<N, 128, 0, stream>>>(x, y, batch, lin_W, lin_b, h, hb, N);
    swizzle_kernel<<<(4 * PER + 255) / 256, 256, 0, stream>>>(c1_Wf, c1_Ws, c2_Wf, c2_Ws, Wsw);

    unsigned short* W1f = Wsw;
    unsigned short* W1s = Wsw + PER;
    unsigned short* W2f = Wsw + 2 * PER;
    unsigned short* W2s = Wsw + 3 * PER;

    int tiles  = (E + 15) / 16;
    int blocks = (tiles + 7) / 8;          // 8 waves / block

    // conv 1 (+ leaky)
    hipMemsetAsync(agg, 0, (size_t)N * CCH * 4, stream);
    cgconv_kernel<<<blocks, 256, 0, stream>>>(src, dst, hb, eattr, W1f, W1s, c1_bf, c1_bs, agg, E);
    finalize_kernel<<<(N * CCH + 255) / 256, 256, 0, stream>>>(h, hb, agg, N * CCH, 1);

    // conv 2 (no activation)
    hipMemsetAsync(agg, 0, (size_t)N * CCH * 4, stream);
    cgconv_kernel<<<blocks, 256, 0, stream>>>(src, dst, hb, eattr, W2f, W2s, c2_bf, c2_bs, agg, E);
    finalize_kernel<<<(N * CCH + 255) / 256, 256, 0, stream>>>(h, hb, agg, N * CCH, 0);

    // mean pool + head
    hipMemsetAsync(sums, 0, (size_t)G * CNC * 4 + (size_t)G * 4, stream);
    pool_kernel<<<((N * CCH) + 255) / 256, 256, 0, stream>>>(h, batch, y, sums, cnt, N, CNC);
    head_kernel<<<G, 32, 0, stream>>>(sums, cnt, fc1_W, fc1_b, fc2_W, fc2_b, (float*)d_out, CNC);
}
